// Vulnerability_5523327943291
// MI455X (gfx1250) — compile-verified, matrix operand order checked by reading, not run on hardware
//
#include <hip/hip_runtime.h>
#include <hip/hip_bf16.h>

typedef __attribute__((ext_vector_type(16))) __bf16 v16bf;
typedef __attribute__((ext_vector_type(8)))  __bf16 v8bf;
typedef __attribute__((ext_vector_type(8)))  float  v8f;

#define TPB 256

// Hardware f32 atomic add, no return, device scope (resolves at L2 atomic
// units -- required for cross-WGP accumulation; WGP scope would be incoherent).
__device__ __forceinline__ void atomic_fadd_dev(float* p, float v) {
    asm volatile("global_atomic_add_f32 %0, %1, off scope:SCOPE_DEV"
                 :: "v"(p), "v"(v) : "memory");
}

// ---------------------------------------------------------------------------
// Degree / normalization:  deg[i] = 1 (self loop) + #edges with dst==i
//                          dinv[i] = rsqrt(deg[i])
// ---------------------------------------------------------------------------
__global__ __launch_bounds__(TPB) void deg_init(float* __restrict__ deg, int n) {
    int i = blockIdx.x * TPB + threadIdx.x;
    if (i < n) deg[i] = 1.0f;                    // self-loop contribution
}

__global__ __launch_bounds__(TPB) void deg_edges(const int* __restrict__ dst,
                                                 float* __restrict__ deg, int e) {
    int i = blockIdx.x * TPB + threadIdx.x;
    if (i < e) atomic_fadd_dev(&deg[dst[i]], 1.0f);
}

__global__ __launch_bounds__(TPB) void deg_rsqrt(float* __restrict__ deg, int n) {
    int i = blockIdx.x * TPB + threadIdx.x;
    if (i < n) deg[i] = rsqrtf(deg[i]);          // deg >= 1 always
}

// ---------------------------------------------------------------------------
// WMMA GEMM with fused GCN prologue/epilogue:
//   Aeff = INRELU ? relu(A + abias[k]) : A          (fused previous-layer bias+relu)
//   H    = Aeff @ W                                  (bf16 WMMA, fp32 acc)
//   Hout = H ;  Agg = H * dinv[row]^2 (+ obias)     (fused self-loop init)
// W pre-swizzled in LDS into exact B-fragment order: per-lane fragment is one
// 32-byte contiguous LDS read (2x ds_load_b128). One wave per 16-row stripe.
// In-place A==Agg is safe: each row is read only by the wave that writes it.
// ---------------------------------------------------------------------------
template <int NCOLS, bool INRELU>
__global__ __launch_bounds__(TPB)
void gcn_gemm_wmma(const float* __restrict__ A, const float* __restrict__ abias,
                   const float* __restrict__ W, const float* __restrict__ dinv,
                   float* __restrict__ Hout, float* __restrict__ Agg,
                   const float* __restrict__ obias, int nrows) {
    constexpr int NTILE = (NCOLS + 15) / 16;       // 8 (D=128) or 1 (OUT=5)
    constexpr int KT    = 4;                       // 128 / 32
    __shared__ v16bf Wf[KT * NTILE * 32];          // 32KB or 4KB of 320KB/WGP

    // Pre-swizzle: unit = (fragIdx, lane, half) writes 8 bf16 (16B) to LDS.
    // Fragment element e = half*8 + j  <->  K = ktile*32 + kb(lane) + half*16 + j
    {
        __bf16* wsw = (__bf16*)Wf;
        for (int unit = threadIdx.x; unit < KT * NTILE * 64; unit += TPB) {
            const int half  = unit & 1;
            const int lane  = (unit >> 1) & 31;
            const int frag  = unit >> 6;           // = ktile*NTILE + tile
            const int tile  = frag % NTILE;
            const int ktile = frag / NTILE;
            const int n     = tile * 16 + (lane & 15);
            const int kbase = ktile * 32 + ((lane < 16) ? 0 : 8) + half * 16;
            v8bf tmp;
#pragma unroll
            for (int j = 0; j < 8; ++j) {
                float v = (n < NCOLS) ? W[(kbase + j) * NCOLS + n] : 0.0f;
                tmp[j] = (__bf16)v;
            }
            *(v8bf*)(wsw + (size_t)unit * 8) = tmp;
        }
    }
    __syncthreads();

    const int wave  = threadIdx.x >> 5;
    const int lane  = threadIdx.x & 31;
    const int mtile = blockIdx.x * (TPB / 32) + wave;
    if (mtile * 16 >= nrows) return;               // wave-uniform: EXEC stays full

    const int kb = (lane < 16) ? 0 : 8;            // per-ISA 16-bit K split
    const long long row = (long long)mtile * 16 + (lane & 15);
    const float* arow = A + row * 128;

    v8f acc[NTILE];
#pragma unroll
    for (int t = 0; t < NTILE; ++t)
#pragma unroll
        for (int r = 0; r < 8; ++r) acc[t][r] = 0.0f;

#pragma unroll
    for (int kt = 0; kt < KT; ++kt) {
        // A fragment: two contiguous 8-float chunks per lane (K = kb+0..7, kb+16..23)
        v16bf afrag;
#pragma unroll
        for (int half = 0; half < 2; ++half) {
            const int kbase = kt * 32 + kb + half * 16;
            float4 q0 = *(const float4*)(arow + kbase);
            float4 q1 = *(const float4*)(arow + kbase + 4);
            if (INRELU) {                          // fused relu(a + b_prev[k])
                const float4 u0 = *(const float4*)(abias + kbase);
                const float4 u1 = *(const float4*)(abias + kbase + 4);
                q0.x = fmaxf(q0.x + u0.x, 0.0f);  q0.y = fmaxf(q0.y + u0.y, 0.0f);
                q0.z = fmaxf(q0.z + u0.z, 0.0f);  q0.w = fmaxf(q0.w + u0.w, 0.0f);
                q1.x = fmaxf(q1.x + u1.x, 0.0f);  q1.y = fmaxf(q1.y + u1.y, 0.0f);
                q1.z = fmaxf(q1.z + u1.z, 0.0f);  q1.w = fmaxf(q1.w + u1.w, 0.0f);
            }
            afrag[half * 8 + 0] = (__bf16)q0.x;
            afrag[half * 8 + 1] = (__bf16)q0.y;
            afrag[half * 8 + 2] = (__bf16)q0.z;
            afrag[half * 8 + 3] = (__bf16)q0.w;
            afrag[half * 8 + 4] = (__bf16)q1.x;
            afrag[half * 8 + 5] = (__bf16)q1.y;
            afrag[half * 8 + 6] = (__bf16)q1.z;
            afrag[half * 8 + 7] = (__bf16)q1.w;
        }
#pragma unroll
        for (int t = 0; t < NTILE; ++t) {
            const v16bf bfrag = Wf[(kt * NTILE + t) * 32 + lane];
            acc[t] = __builtin_amdgcn_wmma_f32_16x16x32_bf16(
                false, afrag, false, bfrag, (short)0, acc[t], false, false);
        }
    }

    // Fused epilogue: Hout = H,  Agg = H * dinv^2 (+obias).
    const int n  = lane & 15;
    const int m0 = (lane < 16) ? 0 : 8;
    float w2[8];
#pragma unroll
    for (int r = 0; r < 8; ++r) {
        float di = dinv[mtile * 16 + m0 + r];
        w2[r] = di * di;
    }
#pragma unroll
    for (int t = 0; t < NTILE; ++t) {
        const int col = t * 16 + n;
        if (col < NCOLS) {
            const float ob = (obias != nullptr) ? obias[col] : 0.0f;
#pragma unroll
            for (int r = 0; r < 8; ++r) {
                const long long orow = (long long)mtile * 16 + m0 + r;
                const float v = acc[t][r];
                Hout[orow * NCOLS + col] = v;
                Agg[orow * NCOLS + col]  = v * w2[r] + ob;
            }
        }
    }
}

// ---------------------------------------------------------------------------
// Edge aggregation, feature dim 128: 32 lanes per edge (wave-uniform indices),
// float4 gather of h[src], 4x hardware f32 atomics into agg[dst].
// ---------------------------------------------------------------------------
__global__ __launch_bounds__(TPB)
void agg_edges_f128(const int* __restrict__ src, const int* __restrict__ dst,
                    const float* __restrict__ dinv, const float* __restrict__ h,
                    float* __restrict__ agg, int nedges) {
    long long t = (long long)blockIdx.x * TPB + threadIdx.x;
    int e    = (int)(t >> 5);          // wave-uniform edge id (32 lanes/edge)
    int lane = (int)(t & 31);
    if (e >= nedges) return;
    int s = src[e], d = dst[e];
    float w = dinv[s] * dinv[d];
    const float4 v = *(const float4*)(h + (long long)s * 128 + lane * 4);
    float* p = agg + (long long)d * 128 + lane * 4;
    atomic_fadd_dev(p + 0, v.x * w);
    atomic_fadd_dev(p + 1, v.y * w);
    atomic_fadd_dev(p + 2, v.z * w);
    atomic_fadd_dev(p + 3, v.w * w);
}

// ---------------------------------------------------------------------------
// Edge aggregation, feature dim 5 (final layer, accumulates into d_out).
// ---------------------------------------------------------------------------
__global__ __launch_bounds__(TPB)
void agg_edges_f5(const int* __restrict__ src, const int* __restrict__ dst,
                  const float* __restrict__ dinv, const float* __restrict__ h5,
                  float* __restrict__ out, int nedges) {
    int e = blockIdx.x * TPB + threadIdx.x;
    if (e >= nedges) return;
    int s = src[e], d = dst[e];
    float w = dinv[s] * dinv[d];
#pragma unroll
    for (int f = 0; f < 5; ++f)
        atomic_fadd_dev(&out[d * 5 + f], h5[s * 5 + f] * w);
}

// ---------------------------------------------------------------------------
extern "C" void kernel_launch(void* const* d_in, const int* in_sizes, int n_in,
                              void* d_out, int out_size, void* d_ws, size_t ws_size,
                              hipStream_t stream) {
    const float* x  = (const float*)d_in[0];
    const int*   ei = (const int*)d_in[1];
    const float* W0 = (const float*)d_in[2];
    const float* b0 = (const float*)d_in[3];
    const float* W1 = (const float*)d_in[4];
    const float* b1 = (const float*)d_in[5];
    const float* W2 = (const float*)d_in[6];
    const float* b2 = (const float*)d_in[7];
    float* out = (float*)d_out;

    const int N = in_sizes[0] / 128;
    const int E = in_sizes[1] / 2;
    const int* src = ei;
    const int* dst = ei + E;

    // Workspace carve-out (256B aligned): dinv[N], h[N*128], agg[N*128]
    char* ws = (char*)d_ws;
    size_t off = 0;
    float* dinv = (float*)(ws + off); off += (((size_t)N * 4 + 255) & ~(size_t)255);
    float* h    = (float*)(ws + off); off += (size_t)N * 128 * 4;
    float* agg  = (float*)(ws + off);

    const int nb_n   = (N + TPB - 1) / TPB;
    const int nb_e   = (E + TPB - 1) / TPB;
    const int nb_e32 = (int)(((long long)E * 32 + TPB - 1) / TPB);
    const int mt     = (N + 15) / 16;
    const int nb_g   = (mt + (TPB / 32) - 1) / (TPB / 32);

    // degree -> dinv
    deg_init<<<nb_n, TPB, 0, stream>>>(dinv, N);
    deg_edges<<<nb_e, TPB, 0, stream>>>(dst, dinv, E);
    deg_rsqrt<<<nb_n, TPB, 0, stream>>>(dinv, N);

    // layer 0: h = x@W0, agg = h*dinv^2 (fused); edges scatter into agg
    gcn_gemm_wmma<128, false><<<nb_g, TPB, 0, stream>>>(
        x, nullptr, W0, dinv, h, agg, nullptr, N);
    agg_edges_f128<<<nb_e32, TPB, 0, stream>>>(src, dst, dinv, h, agg, E);

    // layer 1: A = relu(agg + b0) fused into load; h = A@W1, agg = h*dinv^2
    gcn_gemm_wmma<128, true><<<nb_g, TPB, 0, stream>>>(
        agg, b0, W1, dinv, h, agg, nullptr, N);
    agg_edges_f128<<<nb_e32, TPB, 0, stream>>>(src, dst, dinv, h, agg, E);

    // layer 2: A = relu(agg + b1) fused; h5 = A@W2, out = h5*dinv^2 + b2
    gcn_gemm_wmma<5, true><<<nb_g, TPB, 0, stream>>>(
        agg, b1, W2, dinv, h, out, b2, N);
    agg_edges_f5<<<nb_e, TPB, 0, stream>>>(src, dst, dinv, h, out, E);
}
